// PointNetFeaturePropagation_6803228197528
// MI455X (gfx1250) — matrix operand order repaired and checked by reading, not compiled, not run on hardware
//
#include <hip/hip_runtime.h>
#include <hip/hip_bf16.h>

#define B_    16
#define N_    4096
#define S_    1024
#define D1_   128
#define D2_   256
#define INCH  384
#define OUTCH 128
#define MTOT  (B_ * N_)   // 65536 rows
#define EPS_BN 1e-5f

typedef __attribute__((ext_vector_type(16))) __bf16 v16bf;
typedef __attribute__((ext_vector_type(8)))  float  v8f;
typedef __attribute__((ext_vector_type(4)))  unsigned v4u;
typedef __attribute__((ext_vector_type(8)))  unsigned v8u;
typedef __bf16 bf16;

// ---------------------------------------------------------------------------
// CDNA5 async global->LDS copy (ASYNCcnt-tracked), one 16B chunk per call.
// VDST VGPR carries the LDS byte address (low 32 bits of the generic pointer).
// ---------------------------------------------------------------------------
__device__ __forceinline__ void async_ld_b128(const void* gptr, void* lptr) {
    unsigned           lo = (unsigned)(uintptr_t)lptr;
    unsigned long long ga = (unsigned long long)(uintptr_t)gptr;
    asm volatile("global_load_async_to_lds_b128 %0, %1, off" :: "v"(lo), "v"(ga) : "memory");
}
__device__ __forceinline__ void wait_async0() {
    asm volatile("s_wait_asynccnt 0x0" ::: "memory");
}

// Cooperative async stage of a 128-row x 32-K bf16 tile (8 KB), 256 threads,
// 2 fully-coalesced 16B chunks per thread.
__device__ __forceinline__ void async_copy_tileA(const bf16* __restrict__ A, int K,
                                                 size_t rowTile, int kb,
                                                 bf16* dst, int tid) {
#pragma unroll
    for (int h = 0; h < 2; ++h) {
        int id  = tid + h * 256;       // 0..511 chunk id
        int row = id >> 2;             // 4 x 16B chunks per 64B row
        int c8  = (id & 3) * 8;        // element offset within row
        async_ld_b128(A + (rowTile + row) * (size_t)K + kb + c8, dst + row * 32 + c8);
    }
}

// ---------------------------------------------------------------------------
// Tensor Data Mover: load a 128x128 bf16 row-major tile (32 KB) into LDS with
// one descriptor. D# group0: count=1 | lds_addr | global_addr | type=2.
// group1: data_size=2B, tensor_dim0/1 = 128, tile_dim0/1 = 128, stride0 = 128.
// ---------------------------------------------------------------------------
__device__ __forceinline__ void tdm_load_w128(const bf16* __restrict__ W, bf16* sW) {
    unsigned long long ga  = (unsigned long long)(uintptr_t)W;
    unsigned           lds = (unsigned)(uintptr_t)sW;
    v4u g0 = { 1u,                                     // count=1, user D#
               lds,                                    // lds_addr
               (unsigned)(ga & 0xffffffffu),           // global_addr[31:0]
               (unsigned)((ga >> 32) & 0x01ffffffu) | (2u << 30) }; // ga[56:32] | type=2
    v8u g1 = { (1u << 16),                             // data_size=1 (2 bytes)
               (128u & 0xffffu) << 16,                 // tensor_dim0[15:0] -> [31:16]
               ((128u >> 16) & 0xffffu) | ((128u & 0xffffu) << 16), // dim0 hi | dim1 lo
               ((128u >> 16) & 0xffffu) | (128u << 16),             // dim1 hi | tile_dim0
               128u,                                   // tile_dim1 | tile_dim2=0
               128u,                                   // tensor_dim0_stride[31:0]
               0u,                                     // stride0 hi | stride1 lo
               0u };                                   // stride1 hi
    asm volatile("tensor_load_to_lds %0, %1" :: "s"(g0), "s"(g1) : "memory");
}

// ---------------------------------------------------------------------------
// Fragment loaders for v_wmma_f32_16x16x32_bf16 (wave32).
// ---------------------------------------------------------------------------
__device__ __forceinline__ v16bf frag_from(const bf16* __restrict__ p16a,
                                           const bf16* __restrict__ p16b) {
    union { uint4 q[2]; v16bf v; } u;
    u.q[0] = *reinterpret_cast<const uint4*>(p16a);
    u.q[1] = *reinterpret_cast<const uint4*>(p16b);
    return u.v;
}
// A (16x32 MxK): lanes 0-15 M=lane, V0..3: K=0..7, V4..7: K=16..23; upper half +8.
__device__ __forceinline__ v16bf load_fragA(const bf16* __restrict__ base, int ld,
                                            int row0, int kb, int lane) {
    int r = row0 + (lane & 15);
    int kbase = kb + ((lane >> 4) << 3);
    const bf16* p = base + (size_t)r * ld + kbase;
    return frag_from(p, p + 16);
}
// B (32x16 KxN): lanes 0-15 N=lane, V0..3: K=0..7, V4..7: K=16..23; upper half +8.
__device__ __forceinline__ v16bf load_fragB(const bf16* __restrict__ W, int ld,
                                            int n0, int kb, int lane) {
    int r = n0 + (lane & 15);
    int kbase = kb + ((lane >> 4) << 3);
    const bf16* p = W + (size_t)r * ld + kbase;
    return frag_from(p, p + 16);
}

// ---------------------------------------------------------------------------
// GEMM: Y[m,c] = sum_k A[m,k]*W[c,k] + bias[c]; bf16 in / f32 acc / bf16 out,
// fused per-channel sum & sumsq for batchnorm.
// Block = 8 waves, tile = 128 rows x 128 cols; wave owns a 16-row strip.
// A tiles double-buffered in LDS via async-to-LDS; W in LDS via TDM when K==128.
// ---------------------------------------------------------------------------
__global__ __launch_bounds__(256, 2)
void gemm_bias_stats_kernel(const bf16* __restrict__ A, const bf16* __restrict__ W,
                            const float* __restrict__ bias, bf16* __restrict__ Y,
                            float* __restrict__ stats, int K) {
    __shared__ __align__(16) bf16 sA[2][128 * 32];   // 16 KB double buffer
    __shared__ __align__(16) bf16 sW[OUTCH * OUTCH]; // 32 KB (K==128 path)
    __shared__ float s_sum[OUTCH];
    __shared__ float s_sq[OUTCH];

    const int tid  = threadIdx.x;
    const int wave = tid >> 5;
    const int lane = tid & 31;
    if (tid < OUTCH) { s_sum[tid] = 0.f; s_sq[tid] = 0.f; }

    const size_t rowTile = (size_t)blockIdx.x * 128;
    const int    row0l   = wave * 16;          // strip inside LDS tile
    const int    nk      = K / 32;
    const bool   wInLds  = (K == OUTCH);

    if (wInLds && wave == 0) {
        tdm_load_w128(W, sW);
        __builtin_amdgcn_s_wait_tensorcnt((short)0);
    }

    async_copy_tileA(A, K, rowTile, 0, &sA[0][0], tid);

    v8f acc[8] = {};
    if (wInLds) {
        for (int k = 0; k < nk; ++k) {
            wait_async0();
            __syncthreads();                   // tile k (and sW) visible to all
            if (k + 1 < nk)
                async_copy_tileA(A, K, rowTile, (k + 1) * 32, &sA[(k + 1) & 1][0], tid);
            v16bf a = load_fragA(&sA[k & 1][0], 32, row0l, 0, lane);
            v16bf bfr[8];
#pragma unroll
            for (int g = 0; g < 8; ++g)
                bfr[g] = load_fragB(sW, OUTCH, g * 16, k * 32, lane);   // ds_load
#pragma unroll
            for (int g = 0; g < 8; ++g)
                acc[g] = __builtin_amdgcn_wmma_f32_16x16x32_bf16(
                    false, a, false, bfr[g], (short)0, acc[g], false, false);
        }
    } else {
        for (int k = 0; k < nk; ++k) {
            wait_async0();
            __syncthreads();
            if (k + 1 < nk)
                async_copy_tileA(A, K, rowTile, (k + 1) * 32, &sA[(k + 1) & 1][0], tid);
            v16bf a = load_fragA(&sA[k & 1][0], 32, row0l, 0, lane);
            v16bf bfr[8];
#pragma unroll
            for (int g = 0; g < 8; ++g)
                bfr[g] = load_fragB(W, K, g * 16, k * 32, lane);        // global, L2-hot
#pragma unroll
            for (int g = 0; g < 8; ++g)
                acc[g] = __builtin_amdgcn_wmma_f32_16x16x32_bf16(
                    false, a, false, bfr[g], (short)0, acc[g], false, false);
        }
    }

    // C/D layout: lanes 0-15 -> N=lane, VGPR r -> M=r; lanes 16-31 -> M=r+8.
    const int mOff = (lane < 16) ? 0 : 8;
    const int cl   = lane & 15;
#pragma unroll
    for (int g = 0; g < 8; ++g) {
        int   c  = g * 16 + cl;
        float bv = bias[c];
        float lsum = 0.f, lsq = 0.f;
#pragma unroll
        for (int r = 0; r < 8; ++r) {
            float y = acc[g][r] + bv;
            size_t row = rowTile + row0l + r + mOff;
            Y[row * OUTCH + c] = (bf16)y;
            lsum += y;
            lsq  += y * y;
        }
        atomicAdd(&s_sum[c], lsum);   // ds_add_f32
        atomicAdd(&s_sq[c],  lsq);
    }
    __syncthreads();
    if (tid < OUTCH) {
        atomicAdd(&stats[tid],         s_sum[tid]);
        atomicAdd(&stats[OUTCH + tid], s_sq[tid]);
    }
}

// ---------------------------------------------------------------------------
__global__ void bn_finalize_kernel(const float* __restrict__ stats,
                                   const float* __restrict__ gamma,
                                   const float* __restrict__ beta,
                                   float* __restrict__ sclshift) {
    int c = threadIdx.x;
    if (c < OUTCH) {
        float mean = stats[c] * (1.f / MTOT);
        float var  = stats[OUTCH + c] * (1.f / MTOT) - mean * mean;
        float a    = gamma[c] * rsqrtf(var + EPS_BN);
        sclshift[c]         = a;
        sclshift[OUTCH + c] = beta[c] - mean * a;
    }
}

__global__ __launch_bounds__(256)
void bn_relu_kernel(const bf16* __restrict__ Yin, const float* __restrict__ sclshift,
                    bf16* __restrict__ Xout) {
    size_t i = ((size_t)blockIdx.x * blockDim.x + threadIdx.x) * 4;
    int c = (int)(i & (OUTCH - 1));
#pragma unroll
    for (int k = 0; k < 4; ++k) {
        float y = (float)Yin[i + k];
        float v = y * sclshift[c + k] + sclshift[OUTCH + c + k];
        Xout[i + k] = (bf16)fmaxf(v, 0.f);
    }
}

__global__ void f32_to_bf16_kernel(const float* __restrict__ src,
                                   bf16* __restrict__ dst, int n) {
    int i = blockIdx.x * blockDim.x + threadIdx.x;
    if (i < n) dst[i] = (bf16)src[i];
}

__global__ void zero_kernel(float* __restrict__ p, int n) {
    int i = blockIdx.x * blockDim.x + threadIdx.x;
    if (i < n) p[i] = 0.f;
}

// ---------------------------------------------------------------------------
// points1 (B,128,N) f32 -> feats[:,0:128] bf16 (M x 384), 32x32 LDS transpose.
// ---------------------------------------------------------------------------
__global__ __launch_bounds__(256)
void transpose_p1_kernel(const float* __restrict__ p1, bf16* __restrict__ feats) {
    __shared__ float tile[32][33];
    int b  = blockIdx.z;
    int n0 = blockIdx.x * 32;
    int c0 = blockIdx.y * 32;
    int tx = threadIdx.x, ty = threadIdx.y;   // 32 x 8
#pragma unroll
    for (int j = 0; j < 4; ++j) {
        int c = c0 + ty + j * 8;
        tile[ty + j * 8][tx] = p1[((size_t)b * D1_ + c) * N_ + n0 + tx];
    }
    __syncthreads();
#pragma unroll
    for (int j = 0; j < 4; ++j) {
        int n = n0 + ty + j * 8;
        feats[((size_t)b * N_ + n) * INCH + c0 + tx] = (bf16)tile[tx][ty + j * 8];
    }
}

// ---------------------------------------------------------------------------
// 3-NN + inverse-distance interpolation (see round-1 notes).
// ---------------------------------------------------------------------------
__global__ __launch_bounds__(256)
void knn_interp_kernel(const float* __restrict__ xyz1, const float* __restrict__ xyz2,
                       const float* __restrict__ pts2, bf16* __restrict__ feats) {
    __shared__ float sxyz[S_ * 3];      // 12 KB
    __shared__ int   s_idx[256 * 3];
    __shared__ float s_w[256 * 3];
    int b   = blockIdx.y;
    int n0  = blockIdx.x * 256;
    int tid = threadIdx.x;

    for (int i = tid; i < S_ * 3; i += 256)
        sxyz[i] = xyz2[(size_t)b * S_ * 3 + i];
    __syncthreads();

    int n = n0 + tid;
    float px = xyz1[((size_t)b * N_ + n) * 3 + 0];
    float py = xyz1[((size_t)b * N_ + n) * 3 + 1];
    float pz = xyz1[((size_t)b * N_ + n) * 3 + 2];
    float d0 = 3.4e38f, d1 = 3.4e38f, d2 = 3.4e38f;
    int   i0 = 0, i1 = 0, i2 = 0;
    for (int s = 0; s < S_; ++s) {
        float dx = px - sxyz[s * 3 + 0];
        float dy = py - sxyz[s * 3 + 1];
        float dz = pz - sxyz[s * 3 + 2];
        float d  = dx * dx + dy * dy + dz * dz;
        if (d < d0)      { d2 = d1; i2 = i1; d1 = d0; i1 = i0; d0 = d; i0 = s; }
        else if (d < d1) { d2 = d1; i2 = i1; d1 = d;  i1 = s; }
        else if (d < d2) { d2 = d;  i2 = s; }
    }
    float r0 = 1.f / (d0 + 1e-8f);
    float r1 = 1.f / (d1 + 1e-8f);
    float r2 = 1.f / (d2 + 1e-8f);
    float rs = 1.f / (r0 + r1 + r2);
    s_idx[tid * 3 + 0] = i0; s_idx[tid * 3 + 1] = i1; s_idx[tid * 3 + 2] = i2;
    s_w[tid * 3 + 0] = r0 * rs; s_w[tid * 3 + 1] = r1 * rs; s_w[tid * 3 + 2] = r2 * rs;
    __syncthreads();

    int c = tid;   // thread = channel; coalesced writes, p2 row hot in L0
    const float* p2c = pts2 + ((size_t)b * D2_ + c) * S_;
    for (int p = 0; p < 256; ++p) {
        float v = s_w[p * 3 + 0] * p2c[s_idx[p * 3 + 0]]
                + s_w[p * 3 + 1] * p2c[s_idx[p * 3 + 1]]
                + s_w[p * 3 + 2] * p2c[s_idx[p * 3 + 2]];
        feats[((size_t)b * N_ + n0 + p) * INCH + D1_ + c] = (bf16)v;
    }
}

// ---------------------------------------------------------------------------
// out[b,c,n] = relu(a2[c]*h2 + b2s[c] + x) fused with 32x32 LDS transpose.
// ---------------------------------------------------------------------------
__global__ __launch_bounds__(256)
void epilogue_kernel(const bf16* __restrict__ H2, const bf16* __restrict__ X,
                     const float* __restrict__ sclshift, float* __restrict__ out) {
    __shared__ float tile[32][33];
    int b  = blockIdx.z;
    int n0 = blockIdx.x * 32;
    int c0 = blockIdx.y * 32;
    int tx = threadIdx.x, ty = threadIdx.y;
#pragma unroll
    for (int j = 0; j < 4; ++j) {
        int n = n0 + ty + j * 8;
        size_t row = (size_t)b * N_ + n;
        int c = c0 + tx;
        float h = (float)H2[row * OUTCH + c];
        float x = (float)X[row * OUTCH + c];
        float v = h * sclshift[c] + sclshift[OUTCH + c] + x;
        tile[ty + j * 8][tx] = fmaxf(v, 0.f);
    }
    __syncthreads();
#pragma unroll
    for (int j = 0; j < 4; ++j) {
        int c = c0 + ty + j * 8;
        out[((size_t)b * OUTCH + c) * N_ + n0 + tx] = tile[tx][ty + j * 8];
    }
}

// ---------------------------------------------------------------------------
extern "C" void kernel_launch(void* const* d_in, const int* in_sizes, int n_in,
                              void* d_out, int out_size, void* d_ws, size_t ws_size,
                              hipStream_t stream) {
    const float* xyz1    = (const float*)d_in[0];
    const float* xyz2    = (const float*)d_in[1];
    const float* points1 = (const float*)d_in[2];
    const float* points2 = (const float*)d_in[3];
    const float* fuse_w  = (const float*)d_in[4];
    const float* fuse_b  = (const float*)d_in[5];
    const float* fuse_g  = (const float*)d_in[6];
    const float* fuse_be = (const float*)d_in[7];
    const float* w1      = (const float*)d_in[8];
    const float* b1      = (const float*)d_in[9];
    const float* g1      = (const float*)d_in[10];
    const float* be1     = (const float*)d_in[11];
    const float* w2      = (const float*)d_in[12];
    const float* b2      = (const float*)d_in[13];
    const float* g2      = (const float*)d_in[14];
    const float* be2     = (const float*)d_in[15];
    float* out = (float*)d_out;

    // ---- workspace carve-up (bf16 intermediates; aggressive reuse) ----
    char*  ws    = (char*)d_ws;
    size_t off   = 0;
    bf16*  feats = (bf16*)(ws + off); off += (size_t)MTOT * INCH  * sizeof(bf16); // 48 MB
    bf16*  Y0    = (bf16*)(ws + off); off += (size_t)MTOT * OUTCH * sizeof(bf16); // 16 MB
    bf16*  X     = (bf16*)(ws + off); off += (size_t)MTOT * OUTCH * sizeof(bf16); // 16 MB
    bf16*  H1    = (bf16*)(ws + off); off += (size_t)MTOT * OUTCH * sizeof(bf16); // 16 MB
    bf16*  fwb   = (bf16*)(ws + off); off += (size_t)OUTCH * INCH  * sizeof(bf16);
    bf16*  w1b   = (bf16*)(ws + off); off += (size_t)OUTCH * OUTCH * sizeof(bf16);
    bf16*  w2b   = (bf16*)(ws + off); off += (size_t)OUTCH * OUTCH * sizeof(bf16);
    float* stats = (float*)(ws + off); off += 3 * 2 * OUTCH * sizeof(float);
    float* sclsh = (float*)(ws + off); off += 3 * 2 * OUTCH * sizeof(float);
    bf16*  H1bn  = Y0;     // Y0 dead after first bn_relu
    bf16*  H2    = feats;  // feats dead after GEMM1

    // stats must be zero at the start of every call (graph replays don't re-poison)
    zero_kernel<<<3, 256, 0, stream>>>(stats, 3 * 2 * OUTCH);

    // weights -> bf16
    f32_to_bf16_kernel<<<(OUTCH * INCH + 255) / 256, 256, 0, stream>>>(fuse_w, fwb, OUTCH * INCH);
    f32_to_bf16_kernel<<<(OUTCH * OUTCH + 255) / 256, 256, 0, stream>>>(w1, w1b, OUTCH * OUTCH);
    f32_to_bf16_kernel<<<(OUTCH * OUTCH + 255) / 256, 256, 0, stream>>>(w2, w2b, OUTCH * OUTCH);

    // build fused feature rows: [p1^T | interp]
    transpose_p1_kernel<<<dim3(N_ / 32, D1_ / 32, B_), dim3(32, 8), 0, stream>>>(points1, feats);
    knn_interp_kernel<<<dim3(N_ / 256, B_), 256, 0, stream>>>(xyz1, xyz2, points2, feats);

    // layer 0: fuse (384 -> 128)
    gemm_bias_stats_kernel<<<MTOT / 128, 256, 0, stream>>>(feats, fwb, fuse_b, Y0, stats, INCH);
    bn_finalize_kernel<<<1, 128, 0, stream>>>(stats, fuse_g, fuse_be, sclsh);
    bn_relu_kernel<<<(MTOT * OUTCH / 4) / 256, 256, 0, stream>>>(Y0, sclsh, X);

    // layer 1: 128 -> 128
    gemm_bias_stats_kernel<<<MTOT / 128, 256, 0, stream>>>(X, w1b, b1, H1, stats + 2 * OUTCH, OUTCH);
    bn_finalize_kernel<<<1, 128, 0, stream>>>(stats + 2 * OUTCH, g1, be1, sclsh + 2 * OUTCH);
    bn_relu_kernel<<<(MTOT * OUTCH / 4) / 256, 256, 0, stream>>>(H1, sclsh + 2 * OUTCH, H1bn);

    // layer 2: 128 -> 128 (no relu before residual)
    gemm_bias_stats_kernel<<<MTOT / 128, 256, 0, stream>>>(H1bn, w2b, b2, H2, stats + 4 * OUTCH, OUTCH);
    bn_finalize_kernel<<<1, 128, 0, stream>>>(stats + 4 * OUTCH, g2, be2, sclsh + 4 * OUTCH);

    // out = relu(bn(h2) + x), transposed to (B, 128, N)
    epilogue_kernel<<<dim3(N_ / 32, OUTCH / 32, B_), dim3(32, 8), 0, stream>>>(H2, X, sclsh + 4 * OUTCH, out);
    (void)in_sizes; (void)n_in; (void)out_size; (void)ws_size;
}
